// MultiHeadAttentionLayer_19834158973148
// MI455X (gfx1250) — matrix operand order repaired and checked by reading, not compiled, not run on hardware
//
#include <hip/hip_runtime.h>

#define NODES   20000
#define SEQ     32
#define FEAT    128
#define HDIM    32
#define TILE_N  32
#define NTHR    256
#define LN_EPS  1e-5f
#define RS_H    136   // f16 row stride (272B -> bank-rotating)
#define RS_F    132   // f32 row stride (528B)

typedef __attribute__((ext_vector_type(16))) _Float16 v16h;
typedef __attribute__((ext_vector_type(8)))  _Float16 v8h;
typedef __attribute__((ext_vector_type(8)))  float    v8f;
typedef __attribute__((ext_vector_type(4)))  float    v4f;

struct alignas(16) SharedMem {
  _Float16 a[2][TILE_N * RS_H];        // f16 A staging (key/value/wv), double buffered
  float    qp[TILE_N * RS_F];          // q projection (f32)
  float    pc[TILE_N * RS_F];          // k/v/out projection scratch (f32)
  float    sc[TILE_N * RS_F];          // scores/attn: [n][s*4+h]
  float    bias_q[FEAT], bias_k[FEAT], bias_v[FEAT], bias_o[FEAT];
  float    gam[FEAT], bet[FEAT];
  float    red1[TILE_N][8], red2[TILE_N][8];
  float    mu[TILE_N], rstd[TILE_N];
  int      midx[TILE_N];
};

// B fragment 32x16 of W[kbase..kbase+31][nbase..nbase+15] (matches A K order):
// lane<16 holds B[kbase+{0..7,16..23}][n=lane], lane>=16 holds B[kbase+{8..15,24..31}][n=lane-16].
__device__ inline v16h load_b_frag(const float* __restrict__ W, int kbase, int nbase, int lane) {
  int n  = nbase + (lane & 15);
  int k0 = kbase + ((lane >> 4) << 3);
  v16h b;
#pragma unroll
  for (int i = 0; i < 16; ++i) b[i] = (_Float16)W[(k0 + i + (i >= 8 ? 8 : 0)) * FEAT + n];
  return b;
}

// A fragment 16x32 f16 from LDS: lane<16 -> row=lane, K {0..7,16..23};
// lane>=16 -> row=lane-16, K {8..15,24..31} (per ISA 16-bit A layout).
__device__ inline v16h load_a_frag(const _Float16* __restrict__ lds, int mbase, int kstep, int lane) {
  int row = mbase + (lane & 15);
  int kb  = kstep * 32 + ((lane >> 4) ? 8 : 0);
  const _Float16* p = lds + row * RS_H + kb;
  v8h lo = *(const v8h*)(p);
  v8h hi = *(const v8h*)(p + 16);
  v16h a;
#pragma unroll
  for (int i = 0; i < 8; ++i) { a[i] = lo[i]; a[8 + i] = hi[i]; }
  return a;
}

// 32x128 @ 128x128 slice: this wave computes output columns [16w,16w+16) for
// both 16-row blocks; adds bias, stores f32 result (C/D layout: VGPR i ->
// M = i + 8*(lane>=16), N = lane&15).
__device__ inline void gemm_tile_pair(const _Float16* __restrict__ ldsA, const v16h bfr[4],
                                      float* __restrict__ dst, const float* __restrict__ bias,
                                      int ntile, int lane) {
  v8f c0 = {}; v8f c1 = {};
#pragma unroll
  for (int ks = 0; ks < 4; ++ks) {
    v16h a0 = load_a_frag(ldsA, 0,  ks, lane);
    v16h a1 = load_a_frag(ldsA, 16, ks, lane);
    c0 = __builtin_amdgcn_wmma_f32_16x16x32_f16(false, a0, false, bfr[ks], (short)0, c0, false, false);
    c1 = __builtin_amdgcn_wmma_f32_16x16x32_f16(false, a1, false, bfr[ks], (short)0, c1, false, false);
  }
  int n  = ntile * 16 + (lane & 15);
  int rb = (lane >> 4) ? 8 : 0;
  float bs = bias[n];
#pragma unroll
  for (int i = 0; i < 8; ++i) {
    dst[(rb + i) * RS_F + n]      = c0[i] + bs;
    dst[(16 + rb + i) * RS_F + n] = c1[i] + bs;
  }
}

__device__ inline void load_tile_regs(const float* __restrict__ base, int rn, int fb, v4f r[4]) {
  const float* s = base + (size_t)rn * FEAT + fb;
  r[0] = *(const v4f*)(s); r[1] = *(const v4f*)(s + 4);
  r[2] = *(const v4f*)(s + 8); r[3] = *(const v4f*)(s + 12);
}

__device__ inline void store_tile_f16(_Float16* __restrict__ d0, int rn, int fb, const v4f r[4]) {
  _Float16* d = d0 + rn * RS_H + fb;
#pragma unroll
  for (int i = 0; i < 4; ++i) {
    d[i] = (_Float16)r[0][i]; d[4 + i] = (_Float16)r[1][i];
    d[8 + i] = (_Float16)r[2][i]; d[12 + i] = (_Float16)r[3][i];
  }
}

// masked score for step s: scores[n][s][h] = (q . k)/sqrt(D) + mask
__device__ inline void score_step(SharedMem& sm, int s, int t) {
  if (t < 128) {
    const float scale = 0.17677669529663687f; // 1/sqrt(32)
    int n = t >> 2, h = t & 3;
    const float* kp = sm.pc + n * RS_F + h * HDIM;
    const float* qp = sm.qp + n * RS_F + h * HDIM;
    float acc = 0.f;
#pragma unroll
    for (int d4 = 0; d4 < 8; ++d4) {
      v4f kv = *(const v4f*)(kp + d4 * 4);
      v4f qv = *(const v4f*)(qp + d4 * 4);
      acc += kv[0]*qv[0] + kv[1]*qv[1] + kv[2]*qv[2] + kv[3]*qv[3];
    }
    acc *= scale;
    if (s >= sm.midx[n]) acc -= 1e9f;
    sm.sc[n * RS_F + s * 4 + h] = acc;
  }
}

// attn-weighted accumulation for step s into per-thread wv[16]
__device__ inline void accum_step(SharedMem& sm, int s, int rn, int fb, int hh, float wv[16]) {
  float aw = sm.sc[rn * RS_F + s * 4 + hh];
  const float* vp = sm.pc + rn * RS_F + fb;
#pragma unroll
  for (int i = 0; i < 4; ++i) {
    v4f vv = *(const v4f*)(vp + i * 4);
    wv[i*4+0] += aw * vv[0]; wv[i*4+1] += aw * vv[1];
    wv[i*4+2] += aw * vv[2]; wv[i*4+3] += aw * vv[3];
  }
}

__global__ __launch_bounds__(NTHR) void mha_fused(
    const float* __restrict__ query, const float* __restrict__ key,
    const float* __restrict__ value, const int* __restrict__ mask_idx,
    const float* __restrict__ Wq, const float* __restrict__ bq,
    const float* __restrict__ Wk, const float* __restrict__ bk,
    const float* __restrict__ Wv, const float* __restrict__ bv,
    const float* __restrict__ Wo, const float* __restrict__ bo,
    const float* __restrict__ gamma, const float* __restrict__ beta,
    float* __restrict__ out)
{
  __shared__ SharedMem sm;
  const int t     = threadIdx.x;
  const int lane  = t & 31;
  const int wave  = t >> 5;
  const int node0 = blockIdx.x * TILE_N;
  const int rn    = t >> 3;        // node-in-tile for 16-wide chunk ops
  const int fb    = (t & 7) * 16;  // feature base (within one head block)

  const size_t tile_off = (size_t)node0 * FEAT + (size_t)rn * FEAT + fb;

  // ---- stage constants + query tile (f16) ----
  if (t < FEAT) {
    sm.bias_q[t] = bq[t]; sm.bias_k[t] = bk[t];
    sm.bias_v[t] = bv[t]; sm.bias_o[t] = bo[t];
    sm.gam[t] = gamma[t]; sm.bet[t] = beta[t];
  }
  if (t < TILE_N) sm.midx[t] = mask_idx[node0 + t];
  v4f ra[4], rb[4];
  load_tile_regs(query + (size_t)node0 * FEAT, rn, fb, ra);
  store_tile_f16(sm.a[0], rn, fb, ra);
  __syncthreads();

  // ---- Q projection ----
  {
    v16h bfr[4];
#pragma unroll
    for (int ks = 0; ks < 4; ++ks) bfr[ks] = load_b_frag(Wq, ks * 32, wave * 16, lane);
    gemm_tile_pair(sm.a[0], bfr, sm.qp, sm.bias_q, wave, lane);
  }
  // persistent Wk fragments + key tiles 0,1 (distance-2 staging)
  v16h wkf[4];
#pragma unroll
  for (int ks = 0; ks < 4; ++ks) wkf[ks] = load_b_frag(Wk, ks * 32, wave * 16, lane);
  load_tile_regs(key, rn, fb, ra);                                   // tile 0
  load_tile_regs(key + (size_t)NODES * FEAT, rn, fb, rb);            // tile 1
  __syncthreads();                       // qproj done reading a[0]
  store_tile_f16(sm.a[0], rn, fb, ra);
  __syncthreads();

  // ---- pass 1: K projection + masked scores ----
#pragma unroll 1
  for (int s = 0; s < SEQ; s += 2) {
    // even step: a[0]=tile s, rb=tile s+1; load ra=tile s+2
    if (s + 2 < SEQ) load_tile_regs(key + (size_t)(s + 2) * NODES * FEAT, rn, fb, ra);
    if (s + 4 < SEQ) __builtin_prefetch(key + (size_t)(s + 4) * NODES * FEAT + tile_off, 0, 1);
    gemm_tile_pair(sm.a[0], wkf, sm.pc, sm.bias_k, wave, lane);
    __syncthreads();
    score_step(sm, s, t);
    store_tile_f16(sm.a[1], rn, fb, rb);
    __syncthreads();

    // odd step: a[1]=tile s+1, ra=tile s+2; load rb=tile s+3
    if (s + 3 < SEQ) load_tile_regs(key + (size_t)(s + 3) * NODES * FEAT, rn, fb, rb);
    if (s + 5 < SEQ) __builtin_prefetch(key + (size_t)(s + 5) * NODES * FEAT + tile_off, 0, 1);
    gemm_tile_pair(sm.a[1], wkf, sm.pc, sm.bias_k, wave, lane);
    __syncthreads();
    score_step(sm, s + 1, t);
    if (s + 2 < SEQ) store_tile_f16(sm.a[0], rn, fb, ra);
    __syncthreads();
  }

  // ---- softmax over s (overlap with value tile 0,1 loads) ----
  load_tile_regs(value, rn, fb, ra);                                 // tile 0
  load_tile_regs(value + (size_t)NODES * FEAT, rn, fb, rb);          // tile 1
  if (t < 128) {
    int n = t >> 2, h = t & 3;
    float* sp = sm.sc + n * RS_F + h;
    float m = -3.0e38f;
#pragma unroll
    for (int s2 = 0; s2 < SEQ; ++s2) m = fmaxf(m, sp[s2 * 4]);
    float sum = 0.f;
#pragma unroll
    for (int s2 = 0; s2 < SEQ; ++s2) { float e = __expf(sp[s2 * 4] - m); sp[s2 * 4] = e; sum += e; }
    float inv = 1.f / sum;
#pragma unroll
    for (int s2 = 0; s2 < SEQ; ++s2) sp[s2 * 4] *= inv;
  }
  store_tile_f16(sm.a[0], rn, fb, ra);
  v16h wvf[4];
#pragma unroll
  for (int ks = 0; ks < 4; ++ks) wvf[ks] = load_b_frag(Wv, ks * 32, wave * 16, lane);
  __syncthreads();

  // ---- pass 2: V projection + attn-weighted accumulation ----
  float wv[16];
#pragma unroll
  for (int i = 0; i < 16; ++i) wv[i] = 0.f;
  const int hh = fb >> 5;
#pragma unroll 1
  for (int s = 0; s < SEQ; s += 2) {
    if (s + 2 < SEQ) load_tile_regs(value + (size_t)(s + 2) * NODES * FEAT, rn, fb, ra);
    if (s + 4 < SEQ) __builtin_prefetch(value + (size_t)(s + 4) * NODES * FEAT + tile_off, 0, 1);
    gemm_tile_pair(sm.a[0], wvf, sm.pc, sm.bias_v, wave, lane);
    __syncthreads();
    accum_step(sm, s, rn, fb, hh, wv);
    store_tile_f16(sm.a[1], rn, fb, rb);
    __syncthreads();

    if (s + 3 < SEQ) load_tile_regs(value + (size_t)(s + 3) * NODES * FEAT, rn, fb, rb);
    if (s + 5 < SEQ) __builtin_prefetch(value + (size_t)(s + 5) * NODES * FEAT + tile_off, 0, 1);
    gemm_tile_pair(sm.a[1], wvf, sm.pc, sm.bias_v, wave, lane);
    __syncthreads();
    accum_step(sm, s + 1, rn, fb, hh, wv);
    if (s + 2 < SEQ) store_tile_f16(sm.a[0], rn, fb, ra);
    __syncthreads();
  }

  // ---- output projection ----
  {
    _Float16* d = sm.a[0] + rn * RS_H + fb;
#pragma unroll
    for (int i = 0; i < 16; ++i) d[i] = (_Float16)wv[i];
  }
  v16h wof[4];
#pragma unroll
  for (int ks = 0; ks < 4; ++ks) wof[ks] = load_b_frag(Wo, ks * 32, wave * 16, lane);
  __syncthreads();
  gemm_tile_pair(sm.a[0], wof, sm.pc, sm.bias_o, wave, lane);
  __syncthreads();

  // ---- LayerNorm + residual ----
  {
    const float* op = sm.pc + rn * RS_F + fb;
    float s1 = 0.f, s2v = 0.f;
#pragma unroll
    for (int i = 0; i < 4; ++i) {
      v4f v = *(const v4f*)(op + i * 4);
#pragma unroll
      for (int j = 0; j < 4; ++j) { s1 += v[j]; s2v += v[j] * v[j]; }
    }
    sm.red1[rn][t & 7] = s1; sm.red2[rn][t & 7] = s2v;
  }
  __syncthreads();
  if (t < TILE_N) {
    float s1 = 0.f, s2v = 0.f;
#pragma unroll
    for (int i = 0; i < 8; ++i) { s1 += sm.red1[t][i]; s2v += sm.red2[t][i]; }
    float m   = s1 * (1.0f / FEAT);
    float var = s2v * (1.0f / FEAT) - m * m;
    sm.mu[t] = m; sm.rstd[t] = rsqrtf(var + LN_EPS);
  }
  __syncthreads();
  {
    float m = sm.mu[rn], rs = sm.rstd[rn];
    const float* op = sm.pc + rn * RS_F + fb;
    const float* qi = query + (size_t)(node0 + rn) * FEAT + fb;
    float*       og = out   + (size_t)(node0 + rn) * FEAT + fb;
#pragma unroll
    for (int i = 0; i < 16; i += 4) {
      v4f ov = *(const v4f*)(op + i);
      v4f qv = *(const v4f*)(qi + i);
      v4f res;
#pragma unroll
      for (int j = 0; j < 4; ++j)
        res[j] = qv[j] + (ov[j] - m) * rs * sm.gam[fb + i + j] + sm.bet[fb + i + j];
      *(v4f*)(og + i) = res;
    }
  }
}

extern "C" void kernel_launch(void* const* d_in, const int* in_sizes, int n_in,
                              void* d_out, int out_size, void* d_ws, size_t ws_size,
                              hipStream_t stream) {
  (void)in_sizes; (void)n_in; (void)out_size; (void)d_ws; (void)ws_size;
  const float* query = (const float*)d_in[0];
  const float* key   = (const float*)d_in[1];
  const float* value = (const float*)d_in[2];
  const int*   midx  = (const int*)  d_in[3];
  const float* Wq = (const float*)d_in[4];  const float* bq = (const float*)d_in[5];
  const float* Wk = (const float*)d_in[6];  const float* bk = (const float*)d_in[7];
  const float* Wv = (const float*)d_in[8];  const float* bv = (const float*)d_in[9];
  const float* Wo = (const float*)d_in[10]; const float* bo = (const float*)d_in[11];
  const float* gamma = (const float*)d_in[12];
  const float* beta  = (const float*)d_in[13];
  float* out = (float*)d_out;

  mha_fused<<<NODES / TILE_N, NTHR, 0, stream>>>(
      query, key, value, midx, Wq, bq, Wk, bk, Wv, bv, Wo, bo, gamma, beta, out);
}
